// GAT_56788057588585
// MI455X (gfx1250) — matrix operand order repaired and checked by reading, not compiled, not run on hardware
//
#include <hip/hip_runtime.h>

// ---------------- constants ----------------
#define NN     50000
#define EE     800000
#define ALPHA  0.2f

typedef _Float16 half_t;
typedef __attribute__((ext_vector_type(16))) _Float16 v16h;
typedef __attribute__((ext_vector_type(8)))  _Float16 v8h;
typedef __attribute__((ext_vector_type(8)))  float    v8f;

// ---------------- helpers ----------------
__device__ __forceinline__ void atomicMaxF(float* addr, float v) {
  // ordered-int trick; memory initialized to -inf
  if (v >= 0.0f) atomicMax((int*)addr, __float_as_int(v));
  else           atomicMin((unsigned int*)addr, __float_as_uint(v));
}

// ---------------- elementwise kernels ----------------
__global__ void fill_f32(float* __restrict__ p, float v, int n) {
  int i = blockIdx.x * blockDim.x + threadIdx.x;
  if (i < n) p[i] = v;
}

__global__ void cvt_f32_to_f16(const float* __restrict__ in, half_t* __restrict__ out, int n) {
  int i = blockIdx.x * blockDim.x + threadIdx.x;
  if (i < n) out[i] = (half_t)in[i];
}

__global__ void elu_cvt_f16(const float* __restrict__ in, half_t* __restrict__ out, int n) {
  int i = blockIdx.x * blockDim.x + threadIdx.x;
  if (i < n) {
    float x = in[i];
    float y = (x > 0.0f) ? x : (__expf(x) - 1.0f);
    out[i] = (half_t)y;
  }
}

// ---------------- WMMA GEMM: out[M,O] = A[M,K] @ W[O,K]^T ----------------
// One wave per 16x16 output tile. M (=NN) multiple of 16, K multiple of 32, O multiple of 16.
__global__ __launch_bounds__(32) void gemm_wmma_f16(
    const half_t* __restrict__ A,   // [M, K] fp16
    const half_t* __restrict__ W,   // [O, K] fp16
    float* __restrict__ out,        // [M, O] fp32 (fully overwritten)
    int K, int O)
{
  const int rowTile = blockIdx.x;
  const int colTile = blockIdx.y;
  const int lane    = threadIdx.x;       // 0..31
  const int hsel    = lane >> 4;         // 0: lanes 0-15, 1: lanes 16-31
  const int l15     = lane & 15;

  // CDNA5 16-bit A 16x32 layout: row M = lane&15; lanes 0-15 carry K in [0,8)+[16,24),
  // lanes 16-31 carry K in [8,16)+[24,32).
  const int aRow  = rowTile * 16 + l15;
  const int aBase = hsel * 8;
  // CDNA5 16-bit B 32x16 layout: col N = lane&15; lanes 0-15 carry K in [0,8)+[16,24), lanes 16-31 K in [8,16)+[24,32).
  const int bCol  = colTile * 16 + l15;
  const int bBase = hsel * 8;

  v8f acc = {};
  for (int k0 = 0; k0 < K; k0 += 32) {
    union { v16h v; v8h h[2]; } a, b;
    const half_t* ap = A + (size_t)aRow * K + k0 + aBase;
    a.h[0] = *(const v8h*)(ap);        // K = aBase+0 .. aBase+7
    a.h[1] = *(const v8h*)(ap + 16);   // K = aBase+16 .. aBase+23
    const half_t* bp = W + (size_t)bCol * K + k0 + bBase;
    b.h[0] = *(const v8h*)(bp);        // K = bBase+0 .. bBase+7
    b.h[1] = *(const v8h*)(bp + 16);   // K = bBase+16 .. bBase+23
    acc = __builtin_amdgcn_wmma_f32_16x16x32_f16(
        /*neg_a=*/false, a.v, /*neg_b=*/false, b.v,
        /*c_mod=*/(short)0, acc, /*reuse_a=*/false, /*reuse_b=*/false);
  }

  // C/D layout: element r -> M = rowTile*16 + hsel*8 + r, N = colTile*16 + (lane&15)
  float* op = out + (size_t)(rowTile * 16 + hsel * 8) * O + colTile * 16 + l15;
#pragma unroll
  for (int r = 0; r < 8; ++r) op[(size_t)r * O] = acc[r];
}

// ---------------- attention scalars: a1/a2 = <ft[n,h,:], al/ar[h,:]> ----------------
__global__ void a12_kernel(const float* __restrict__ ft, const float* __restrict__ al,
                           const float* __restrict__ ar, float* __restrict__ a1,
                           float* __restrict__ a2, int n, int H, int D) {
  int t = blockIdx.x * blockDim.x + threadIdx.x;
  if (t >= n * H) return;
  int node = t / H, h = t % H;
  const float* f   = ft + (size_t)node * H * D + h * D;
  const float* alh = al + h * D;
  const float* arh = ar + h * D;
  float s1 = 0.0f, s2 = 0.0f;
  for (int d = 0; d < D; ++d) { s1 += f[d] * alh[d]; s2 += f[d] * arh[d]; }
  a1[t] = s1;
  a2[t] = s2;
}

// ---------------- edge pass 1: leaky-relu scores + segment max by dst ----------------
__global__ void edge_pass1(const int* __restrict__ src, const int* __restrict__ dst,
                           const float* __restrict__ a1, const float* __restrict__ a2,
                           float* __restrict__ ebuf, float* __restrict__ emax,
                           int E, int H) {
  int i = blockIdx.x * blockDim.x + threadIdx.x;
  if (i >= E) return;
  int s = src[i], d = dst[i];
  for (int h = 0; h < H; ++h) {
    float e = a1[s * H + h] + a2[d * H + h];
    e = (e > 0.0f) ? e : ALPHA * e;
    ebuf[(size_t)i * H + h] = e;
    atomicMaxF(&emax[d * H + h], e);
  }
}

// ---------------- edge pass 2: exp(e - max) + segment sum ----------------
__global__ void edge_pass2(const int* __restrict__ dst, float* __restrict__ ebuf,
                           const float* __restrict__ emax, float* __restrict__ esum,
                           int E, int H) {
  int t = blockIdx.x * blockDim.x + threadIdx.x;
  if (t >= E * H) return;
  int i = t / H, h = t % H;
  int d = dst[i];
  float ee = __expf(ebuf[t] - emax[d * H + h]);
  ebuf[t] = ee;
  atomicAdd(&esum[d * H + h], ee);
}

// ---------------- edge pass 3: normalize + gather-multiply-scatter ----------------
__global__ void edge_pass3(const int* __restrict__ src, const int* __restrict__ dst,
                           const float* __restrict__ ebuf, const float* __restrict__ esum,
                           const float* __restrict__ ft, float* __restrict__ out,
                           int E, int H, int D) {
  int t = blockIdx.x * blockDim.x + threadIdx.x;
  if (t >= E * H) return;
  int i = t / H, h = t % H;
  int s = src[i], d = dst[i];
  float att = ebuf[t] / esum[d * H + h];
  const float* f = ft + (size_t)s * H * D + h * D;
  float*       o = out + (size_t)d * H * D + h * D;
  for (int dd = 0; dd < D; ++dd) atomicAdd(&o[dd], f[dd] * att);
}

// ---------------- host-side orchestration ----------------
static inline int cdiv(int a, int b) { return (a + b - 1) / b; }

extern "C" void kernel_launch(void* const* d_in, const int* in_sizes, int n_in,
                              void* d_out, int out_size, void* d_ws, size_t ws_size,
                              hipStream_t stream) {
  (void)in_sizes; (void)n_in; (void)out_size; (void)ws_size;

  const float* inputs = (const float*)d_in[0];
  const int*   src    = (const int*)d_in[1];
  const int*   dst    = (const int*)d_in[2];
  const float* W0     = (const float*)d_in[3];
  const float* al0    = (const float*)d_in[4];
  const float* ar0    = (const float*)d_in[5];
  const float* W1     = (const float*)d_in[6];
  const float* al1    = (const float*)d_in[7];
  const float* ar1    = (const float*)d_in[8];
  const float* Wres1  = (const float*)d_in[9];
  const float* W2     = (const float*)d_in[10];
  const float* al2    = (const float*)d_in[11];
  const float* ar2    = (const float*)d_in[12];
  const float* Wres2  = (const float*)d_in[13];

  float* outp = (float*)d_out;        // [NN, 16]

  // workspace layout (all offsets 256B-aligned)
  char* ws = (char*)d_ws;
  float*  ftbuf  = (float*)ws;  ws += (size_t)NN * 128 * 4;   // ft: [N,H,D]
  float*  outbuf = (float*)ws;  ws += (size_t)NN * 128 * 4;   // layer output accumulator
  half_t* h16    = (half_t*)ws; ws += (size_t)NN * 128 * 2;   // fp16 activations
  float*  a1b    = (float*)ws;  ws += (size_t)NN * 4 * 4;
  float*  a2b    = (float*)ws;  ws += (size_t)NN * 4 * 4;
  float*  emaxb  = (float*)ws;  ws += (size_t)NN * 4 * 4;
  float*  esumb  = (float*)ws;  ws += (size_t)NN * 4 * 4;
  float*  ebuf   = (float*)ws;  ws += (size_t)EE * 4 * 4;
  half_t* w16a   = (half_t*)ws; ws += (size_t)128 * 128 * 2;
  half_t* w16b   = (half_t*)ws; ws += (size_t)128 * 128 * 2;

  const int B = 256;
  const int rowTiles = NN / 16;  // 3125, exact
  const float NEG_INF = -__builtin_huge_valf();

  // ================= Layer 0: IN=64 -> H=4 x D=32, no residual =================
  cvt_f32_to_f16<<<cdiv(NN * 64, B), B, 0, stream>>>(inputs, h16, NN * 64);
  cvt_f32_to_f16<<<cdiv(128 * 64, B), B, 0, stream>>>(W0, w16a, 128 * 64);
  gemm_wmma_f16<<<dim3(rowTiles, 8), 32, 0, stream>>>(h16, w16a, ftbuf, 64, 128);
  fill_f32<<<cdiv(NN * 128, B), B, 0, stream>>>(outbuf, 0.0f, NN * 128);
  a12_kernel<<<cdiv(NN * 4, B), B, 0, stream>>>(ftbuf, al0, ar0, a1b, a2b, NN, 4, 32);
  fill_f32<<<cdiv(NN * 4, B), B, 0, stream>>>(emaxb, NEG_INF, NN * 4);
  fill_f32<<<cdiv(NN * 4, B), B, 0, stream>>>(esumb, 0.0f, NN * 4);
  edge_pass1<<<cdiv(EE, B), B, 0, stream>>>(src, dst, a1b, a2b, ebuf, emaxb, EE, 4);
  edge_pass2<<<cdiv(EE * 4, B), B, 0, stream>>>(dst, ebuf, emaxb, esumb, EE, 4);
  edge_pass3<<<cdiv(EE * 4, B), B, 0, stream>>>(src, dst, ebuf, esumb, ftbuf, outbuf, EE, 4, 32);
  elu_cvt_f16<<<cdiv(NN * 128, B), B, 0, stream>>>(outbuf, h16, NN * 128);

  // ================= Layer 1: 128 -> H=4 x D=32, residual =================
  cvt_f32_to_f16<<<cdiv(128 * 128, B), B, 0, stream>>>(W1, w16a, 128 * 128);
  cvt_f32_to_f16<<<cdiv(128 * 128, B), B, 0, stream>>>(Wres1, w16b, 128 * 128);
  gemm_wmma_f16<<<dim3(rowTiles, 8), 32, 0, stream>>>(h16, w16b, outbuf, 128, 128); // residual init
  gemm_wmma_f16<<<dim3(rowTiles, 8), 32, 0, stream>>>(h16, w16a, ftbuf, 128, 128);
  a12_kernel<<<cdiv(NN * 4, B), B, 0, stream>>>(ftbuf, al1, ar1, a1b, a2b, NN, 4, 32);
  fill_f32<<<cdiv(NN * 4, B), B, 0, stream>>>(emaxb, NEG_INF, NN * 4);
  fill_f32<<<cdiv(NN * 4, B), B, 0, stream>>>(esumb, 0.0f, NN * 4);
  edge_pass1<<<cdiv(EE, B), B, 0, stream>>>(src, dst, a1b, a2b, ebuf, emaxb, EE, 4);
  edge_pass2<<<cdiv(EE * 4, B), B, 0, stream>>>(dst, ebuf, emaxb, esumb, EE, 4);
  edge_pass3<<<cdiv(EE * 4, B), B, 0, stream>>>(src, dst, ebuf, esumb, ftbuf, outbuf, EE, 4, 32);
  elu_cvt_f16<<<cdiv(NN * 128, B), B, 0, stream>>>(outbuf, h16, NN * 128);

  // ================= Layer 2: 128 -> 1 head x C=16, residual, output to d_out ======
  cvt_f32_to_f16<<<cdiv(16 * 128, B), B, 0, stream>>>(W2, w16a, 16 * 128);
  cvt_f32_to_f16<<<cdiv(16 * 128, B), B, 0, stream>>>(Wres2, w16b, 16 * 128);
  gemm_wmma_f16<<<dim3(rowTiles, 1), 32, 0, stream>>>(h16, w16b, outp, 128, 16);   // residual init
  gemm_wmma_f16<<<dim3(rowTiles, 1), 32, 0, stream>>>(h16, w16a, ftbuf, 128, 16);
  a12_kernel<<<cdiv(NN * 1, B), B, 0, stream>>>(ftbuf, al2, ar2, a1b, a2b, NN, 1, 16);
  fill_f32<<<cdiv(NN * 1, B), B, 0, stream>>>(emaxb, NEG_INF, NN * 1);
  fill_f32<<<cdiv(NN * 1, B), B, 0, stream>>>(esumb, 0.0f, NN * 1);
  edge_pass1<<<cdiv(EE, B), B, 0, stream>>>(src, dst, a1b, a2b, ebuf, emaxb, EE, 1);
  edge_pass2<<<cdiv(EE * 1, B), B, 0, stream>>>(dst, ebuf, emaxb, esumb, EE, 1);
  edge_pass3<<<cdiv(EE * 1, B), B, 0, stream>>>(src, dst, ebuf, esumb, ftbuf, outp, EE, 1, 16);
}
